// PerformerSeperator_29308856828124
// MI455X (gfx1250) — matrix operand order repaired and checked by reading, not compiled, not run on hardware
//
#include <hip/hip_runtime.h>
#include <stdint.h>

#define DEV __device__ __forceinline__

typedef unsigned short u16;
typedef unsigned int   u32;
typedef __attribute__((ext_vector_type(16))) __bf16 v16bf;
typedef __attribute__((ext_vector_type(8)))  float  v8f;
typedef __attribute__((ext_vector_type(4)))  unsigned int u32x4;
typedef __attribute__((ext_vector_type(8)))  int i32x8;
typedef __attribute__((ext_vector_type(4)))  int i32x4;

static constexpr int BZ = 32, T = 512, DIM = 256, DEPTH = 6, HEADS = 8, NBF = 128, DH = 32, FF = 1024;
static constexpr int QKV  = 3 * DIM;                  // fused q|k|v width = 768
static constexpr int ROWS = BZ * T;                   // 16384 token rows
static constexpr float DN    = 0.42044820762685725f;  // 32^-0.25
static constexpr float RATIO = 0.08838834764831845f;  // 128^-0.5
static constexpr float KEPS  = 1e-4f;

DEV u16  f2bf(float f) { u32 u = __float_as_uint(f); return (u16)((u + 0x7FFFu + ((u >> 16) & 1u)) >> 16); }
DEV float bf2f(u16 h)  { return __uint_as_float(((u32)h) << 16); }
DEV u32  encf(float f) { u32 u = __float_as_uint(f); return (u & 0x80000000u) ? ~u : (u | 0x80000000u); }
DEV float decf(u32 e)  { u32 u = (e & 0x80000000u) ? (e ^ 0x80000000u) : ~e; return __uint_as_float(u); }
DEV float gelu_f(float x) { return 0.5f * x * (1.f + tanhf(0.7978845608028654f * (x + 0.044715f * x * x * x))); }
DEV float wsum(float v)  { for (int o = 16; o; o >>= 1) v += __shfl_xor(v, o); return v; }
DEV float wmaxr(float v) { for (int o = 16; o; o >>= 1) v = fmaxf(v, __shfl_xor(v, o)); return v; }

union ABf { v16bf v; u32 u[8]; uint4 q[2]; };

// ---------------------------------------------------------------------------
// Batched bf16 WMMA GEMM. B is pre-packed in WMMA-fragment order; the block's
// whole B panel (K/32 rows x NW KB, row stride nT16 KB) is staged into LDS
// once — by the Tensor Data Mover when available, else a cooperative copy —
// then the K loop runs software-pipelined: global b128 A loads + ds_load_b128
// B fragments + v_wmma. wave -> 16 x (16*NW) tile; 8 waves/block -> 128 rows.
// K must be a multiple of 64. Fused bias / rowScale / gelu / residual.
// Batch offset = (z/batchH)*Outer + (z%batchH)*Inner.
// ---------------------------------------------------------------------------
template <int NW>
__global__ __launch_bounds__(256) void gemm_pk(
    const u16* __restrict__ A, const u32* __restrict__ Bp,
    float* __restrict__ outF, u16* __restrict__ outB,
    const float* __restrict__ bias, const float* __restrict__ resid,
    const float* __restrict__ rowScale, int rsStride,
    int K, int lda, int nT16, int ldc, int batchH,
    long long aO, long long aI, long long bO, long long bI,
    long long cO, long long cI, int doGelu)
{
  extern __shared__ __align__(16) u32 smem[];   // (K/32) * NW * 256 dwords

  const int z  = blockIdx.z;
  const int zb = z / batchH, zh = z % batchH;
  A  += zb * aO + zh * aI;
  Bp += zb * bO + zh * bI;
  const long long cOff = (long long)zb * cO + (long long)zh * cI;

  const int wave = threadIdx.x >> 5;
  const int lane = threadIdx.x & 31;
  const int row0   = blockIdx.y * 128 + wave * 16;
  const int ntile0 = blockIdx.x * NW;
  const int mrow = lane & 15;   // A: M row / B,C: N col
  const int hs   = lane >> 4;   // half-select

  // ---- stage packed B panel into LDS ----
  const u32* bpanel = Bp + (long long)ntile0 * 256;
#if __has_builtin(__builtin_amdgcn_tensor_load_to_lds) && __has_builtin(__builtin_amdgcn_s_wait_tensorcnt)
  if (wave == 0) {
    const unsigned long long ga = (unsigned long long)(uintptr_t)bpanel;
    const u32 ldsA  = (u32)(uintptr_t)&smem[0];
    const u32 tile0 = (u32)(NW * 128);          // row length, 8-byte units
    const u32 tile1 = (u32)(K >> 5);            // number of rows
    const unsigned long long st0 = (unsigned long long)nT16 * 128;  // row stride, 8B units
    const u32 td0 = 0x40000000u, td1 = 0x40000000u;  // huge tensor dims: no clipping
    u32x4 g0 = { 1u,                              // count=1, user descriptor
                 ldsA,                            // lds_addr
                 (u32)ga,                         // global_addr[31:0]
                 (u32)((ga >> 32) & 0x01FFFFFFu) | (2u << 30) };  // addr[56:32] | type=2
    i32x8 g1 = { (int)(3u << 16),                               // wg_mask=0, data_size=8B
                 (int)(td0 << 16),                              // tensor_dim0[15:0]
                 (int)((td0 >> 16) | (td1 << 16)),              // td0 hi | td1 lo
                 (int)((td1 >> 16) | (tile0 << 16)),            // td1 hi | tile_dim0
                 (int)tile1,                                    // tile_dim1 | tile_dim2=0
                 (int)(u32)(st0 & 0xFFFFFFFFu),                 // dim0_stride[31:0]
                 (int)(u32)(st0 >> 32),                         // stride hi | dim1_stride lo
                 0 };
    i32x4 g2 = { 0, 0, 0, 0 };
    i32x4 g3 = { 0, 0, 0, 0 };
    i32x8 g4 = { 0, 0, 0, 0, 0, 0, 0, 0 };
    __builtin_amdgcn_tensor_load_to_lds(g0, g1, g2, g3, g4, 0);
    __builtin_amdgcn_s_wait_tensorcnt((short)0);
  }
#else
  {
    const int nDw = (K >> 5) * NW * 256;
    const int rowDw = NW * 256;
    for (int i = threadIdx.x * 4; i < nDw; i += 256 * 4) {
      const int r = i / rowDw, wi = i % rowDw;
      *(uint4*)&smem[i] = *(const uint4*)(bpanel + (long long)r * nT16 * 256 + wi);
    }
  }
#endif
  __syncthreads();

  v8f acc[NW];
  #pragma unroll
  for (int t = 0; t < NW; t++)
    #pragma unroll
    for (int r = 0; r < 8; r++) acc[t][r] = 0.f;

  const u16* arow = A + (long long)(row0 + mrow) * lda;

  auto loadA = [&](ABf& a, int k) {
    const u16* ap = arow + k + hs * 16;
    a.q[0] = *(const uint4*)(ap);        // K = k+hs*16   .. +7
    a.q[1] = *(const uint4*)(ap + 8);    // K = k+hs*16+8 .. +7
  };
  auto loadB = [&](ABf* b, int k) {
    #pragma unroll
    for (int t = 0; t < NW; t++) {
      const u32* bp = smem + (((k >> 5) * NW + t) * 32 + lane) * 8;
      b[t].q[0] = *(const uint4*)(bp);
      b[t].q[1] = *(const uint4*)(bp + 4);
    }
  };

  ABf a0, a1, b0[NW], b1[NW];
  loadA(a0, 0);
  loadB(b0, 0);
  for (int k0 = 0; k0 < K; k0 += 64) {
    loadA(a1, k0 + 32);
    loadB(b1, k0 + 32);
    #pragma unroll
    for (int t = 0; t < NW; t++)
      acc[t] = __builtin_amdgcn_wmma_f32_16x16x32_bf16(false, a0.v, false, b0[t].v,
                                                       (short)0, acc[t], false, false);
    const int kn = (k0 + 64 < K) ? k0 + 64 : 0;  // clamped (redundant) load on last iter
    loadA(a0, kn);
    loadB(b0, kn);
    #pragma unroll
    for (int t = 0; t < NW; t++)
      acc[t] = __builtin_amdgcn_wmma_f32_16x16x32_bf16(false, a1.v, false, b1[t].v,
                                                       (short)0, acc[t], false, false);
  }

  #pragma unroll
  for (int t = 0; t < NW; t++) {
    const int col = (ntile0 + t) * 16 + mrow;
    const float bi = bias ? bias[col] : 0.f;
    #pragma unroll
    for (int r = 0; r < 8; r++) {
      const int row = row0 + r + 8 * hs;
      float v = acc[t][r] + bi;
      if (rowScale) v *= rowScale[(long long)z * rsStride + row];
      if (doGelu)   v = gelu_f(v);
      const long long idx = cOff + (long long)row * ldc + col;
      if (resid) v += resid[idx];
      if (outF) outF[idx] = v;
      if (outB) outB[idx] = f2bf(v);
    }
  }
}

// --------------------------- repack kernels --------------------------------

// row-major bf16 (K x N, ldb) -> packed WMMA-B fragments; batched
__global__ __launch_bounds__(256) void repack_b16(
    const u16* __restrict__ in, u32* __restrict__ outp,
    int nT16, int ldb, int batchH,
    long long iO, long long iI, long long oStride, int nDw) {
  const int z = blockIdx.y;
  const u16* ip = in + (long long)(z / batchH) * iO + (long long)(z % batchH) * iI;
  u32* op = outp + (long long)z * oStride;
  const int i = blockIdx.x * 256 + threadIdx.x;
  if (i >= nDw) return;
  const int j = i & 7, l = (i >> 3) & 31, f = i >> 8;
  const int kt2 = f / nT16, nt = f % nT16;
  const int kk = kt2 * 32 + (l >> 4) * 16 + 2 * j;
  const int n  = nt * 16 + (l & 15);
  u32 lo = ip[(long long)kk * ldb + n];
  u32 hi = ip[(long long)(kk + 1) * ldb + n];
  op[i] = lo | (hi << 16);
}

// f32 weights (per-layer contiguous K x N) -> packed bf16 fragments with an
// output N-tile offset (for fusing several weight matrices side by side).
__global__ __launch_bounds__(256) void repack_w(
    const float* __restrict__ in, u32* __restrict__ outp,
    int nT16, int N, int nDw, int outT16, int ntOff, long long oLayer) {
  const int z = blockIdx.y; // layer
  const float* ip = in + (long long)z * nDw * 2;
  u32* op = outp + (long long)z * oLayer;
  const int i = blockIdx.x * 256 + threadIdx.x;
  if (i >= nDw) return;
  const int j = i & 7, l = (i >> 3) & 31, f = i >> 8;
  const int kt2 = f / nT16, nt = f % nT16;
  const int kk = kt2 * 32 + (l >> 4) * 16 + 2 * j;
  const int n  = nt * 16 + (l & 15);
  u32 lo = f2bf(ip[(long long)kk * N + n]);
  u32 hi = f2bf(ip[(long long)(kk + 1) * N + n]);
  op[(((long long)kt2 * outT16 + ntOff + nt) * 32 + l) * 8 + j] = lo | (hi << 16);
}

// --------------------------- helper kernels --------------------------------

__global__ __launch_bounds__(256) void transpose_kernel(const float* __restrict__ mel,
                                                        float* __restrict__ x) {
  __shared__ float tile[32][33];
  const int b = blockIdx.z;
  const int d0 = blockIdx.y * 32, t0 = blockIdx.x * 32;
  const int lx = threadIdx.x & 31, ly = threadIdx.x >> 5;
  #pragma unroll
  for (int r = 0; r < 32; r += 8)
    tile[ly + r][lx] = mel[((long long)b * DIM + d0 + ly + r) * T + t0 + lx];
  __syncthreads();
  #pragma unroll
  for (int r = 0; r < 32; r += 8)
    x[((long long)b * T + t0 + ly + r) * DIM + d0 + lx] = tile[lx][ly + r];
}

// wave per row: LayerNorm -> bf16
__global__ __launch_bounds__(256) void ln_kernel(const float* __restrict__ x,
                                                 const float* __restrict__ g,
                                                 const float* __restrict__ b,
                                                 u16* __restrict__ o) {
  const int wave = threadIdx.x >> 5, lane = threadIdx.x & 31;
  const long long row = (long long)blockIdx.x * 8 + wave;
  const float* xr = x + row * DIM;
  float v[8]; float s = 0.f;
  #pragma unroll
  for (int j = 0; j < 8; j++) { v[j] = xr[lane + 32 * j]; s += v[j]; }
  s = wsum(s);
  const float mu = s * (1.f / 256.f);
  float q = 0.f;
  #pragma unroll
  for (int j = 0; j < 8; j++) { float d = v[j] - mu; q += d * d; }
  q = wsum(q);
  const float rstd = rsqrtf(q * (1.f / 256.f) + 1e-5f);
  u16* orow = o + row * DIM;
  #pragma unroll
  for (int j = 0; j < 8; j++) {
    const int c = lane + 32 * j;
    orow[c] = f2bf((v[j] - mu) * rstd * g[c] + b[c]);
  }
}

// FAVOR+ query features: wave per (b,h,t); per-row max. Reads q from fused qkv.
__global__ __launch_bounds__(256) void qfeat_kernel(const u16* __restrict__ qkv,
                                                    const float* __restrict__ proj,
                                                    u16* __restrict__ qp) {
  const int wave = threadIdx.x >> 5, lane = threadIdx.x & 31;
  const long long idx = (long long)blockIdx.x * 8 + wave; // = bh*T + t
  const int t = (int)(idx & (T - 1));
  const int bh = (int)(idx >> 9), b = bh >> 3, h = bh & 7;
  __shared__ float rowb[8][32];
  const float qv = bf2f(qkv[((long long)b * T + t) * QKV + h * DH + lane]) * DN;
  rowb[wave][lane] = qv;
  __syncthreads();
  const float dg = 0.5f * wsum(qv * qv);
  float xdv[4]; float mx = -3.0e38f;
  #pragma unroll
  for (int j = 0; j < 4; j++) {
    const int m = lane + 32 * j;
    const float* pr = proj + m * DH;
    float s = 0.f;
    #pragma unroll
    for (int d = 0; d < DH; d++) s += rowb[wave][d] * pr[d];
    xdv[j] = s; mx = fmaxf(mx, s);
  }
  mx = wmaxr(mx);
  u16* qr = qp + idx * NBF;
  #pragma unroll
  for (int j = 0; j < 4; j++)
    qr[lane + 32 * j] = f2bf(RATIO * (expf(xdv[j] - dg - mx) + KEPS));
}

// FAVOR+ key features pass 1: xd + diag + per-(b,h) max. Reads k from fused qkv.
__global__ __launch_bounds__(256) void kfeat1_kernel(const u16* __restrict__ qkv,
                                                     const float* __restrict__ proj,
                                                     float* __restrict__ xd,
                                                     float* __restrict__ diag,
                                                     u32* __restrict__ maxbuf) {
  const int wave = threadIdx.x >> 5, lane = threadIdx.x & 31;
  const long long idx = (long long)blockIdx.x * 8 + wave;
  const int t = (int)(idx & (T - 1));
  const int bh = (int)(idx >> 9), b = bh >> 3, h = bh & 7;
  __shared__ float rowb[8][32];
  __shared__ float wmx[8];
  const float kv = bf2f(qkv[((long long)b * T + t) * QKV + DIM + h * DH + lane]) * DN;
  rowb[wave][lane] = kv;
  __syncthreads();
  const float dg = 0.5f * wsum(kv * kv);
  float mx = -3.0e38f;
  float* xdr = xd + idx * NBF;
  #pragma unroll
  for (int j = 0; j < 4; j++) {
    const int m = lane + 32 * j;
    const float* pr = proj + m * DH;
    float s = 0.f;
    #pragma unroll
    for (int d = 0; d < DH; d++) s += rowb[wave][d] * pr[d];
    xdr[m] = s; mx = fmaxf(mx, s);
  }
  if (lane == 0) diag[idx] = dg;
  mx = wmaxr(mx);
  if (lane == 0) wmx[wave] = mx;
  __syncthreads();
  if (threadIdx.x == 0) {
    float bm2 = wmx[0];
    #pragma unroll
    for (int i = 1; i < 8; i++) bm2 = fmaxf(bm2, wmx[i]);
    atomicMax(&maxbuf[bh], encf(bm2));
  }
}

// key features pass 2: kp (transposed, bf16) + kpsum over T
__global__ __launch_bounds__(128) void kfeat2_kernel(const float* __restrict__ xd,
                                                     const float* __restrict__ diag,
                                                     const u32* __restrict__ maxbuf,
                                                     u16* __restrict__ kpT,
                                                     float* __restrict__ kpsum) {
  const int bh = blockIdx.x;
  const int m = threadIdx.x;
  const float mx = decf(maxbuf[bh]);
  const float* xdp = xd + (long long)bh * T * NBF;
  const float* dg = diag + (long long)bh * T;
  u16* kt = kpT + (long long)bh * NBF * T + (long long)m * T;
  float s = 0.f;
  for (int t = 0; t < T; t++) {
    float v = RATIO * (expf(xdp[(long long)t * NBF + m] - dg[t] - mx) + KEPS);
    kt[t] = f2bf(v);
    s += v;
  }
  kpsum[bh * NBF + m] = s;
}

// denom^-1 per (b,h,t): 1 / (qp_row . kpsum)
__global__ __launch_bounds__(256) void denom_kernel(const u16* __restrict__ qp,
                                                    const float* __restrict__ kpsum,
                                                    float* __restrict__ dinv) {
  const long long i = (long long)blockIdx.x * 256 + threadIdx.x; // over B*H*T
  const int bh = (int)(i >> 9);
  const u16* qr = qp + i * NBF;
  const float* ks = kpsum + bh * NBF;
  float s = 0.f;
  for (int m = 0; m < NBF; m++) s += bf2f(qr[m]) * ks[m];
  dinv[i] = 1.f / s;
}

// final masks: out[b, mk, t] = x[b,t,:] . Wm[:,mk] + bm[mk]
__global__ __launch_bounds__(256) void mask_kernel(const float* __restrict__ x,
                                                   const float* __restrict__ Wm,
                                                   const float* __restrict__ bm,
                                                   float* __restrict__ out) {
  const int i = blockIdx.x * 256 + threadIdx.x; // over B*4*T = 65536
  const int t = i & (T - 1);
  const int mk = (i >> 9) & 3;
  const int b = i >> 11;
  const float* xr = x + ((long long)b * T + t) * DIM;
  float s = bm[mk];
  for (int d = 0; d < DIM; d++) s += xr[d] * Wm[d * 4 + mk];
  out[i] = s;
}

// ---------------------------------------------------------------------------

extern "C" void kernel_launch(void* const* d_in, const int* in_sizes, int n_in,
                              void* d_out, int out_size, void* d_ws, size_t ws_size,
                              hipStream_t stream) {
  (void)in_sizes; (void)n_in; (void)out_size; (void)ws_size;
  const float* mel  = (const float*)d_in[0];
  const float* proj = (const float*)d_in[1];
  const float* ln1g = (const float*)d_in[2];
  const float* ln1b = (const float*)d_in[3];
  const float* Wq   = (const float*)d_in[4];
  const float* Wk   = (const float*)d_in[5];
  const float* Wv   = (const float*)d_in[6];
  const float* Wo   = (const float*)d_in[7];
  const float* bo   = (const float*)d_in[8];
  const float* ln2g = (const float*)d_in[9];
  const float* ln2b = (const float*)d_in[10];
  const float* W1   = (const float*)d_in[11];
  const float* b1   = (const float*)d_in[12];
  const float* W2   = (const float*)d_in[13];
  const float* b2   = (const float*)d_in[14];
  const float* Wm   = (const float*)d_in[15];
  const float* bm   = (const float*)d_in[16];
  float* out = (float*)d_out;

  char* w = (char*)d_ws;
  size_t off = 0;
  auto alloc = [&](size_t bytes) -> char* {
    char* p = w + off; off = (off + bytes + 255) & ~(size_t)255; return p;
  };
  float* xf     = (float*)alloc((size_t)ROWS * DIM * 4);
  u16*   hb     = (u16*)  alloc((size_t)ROWS * DIM * 2);
  u16*   qkv    = (u16*)  alloc((size_t)ROWS * QKV * 2);
  u32*   vpk    = (u32*)  alloc((size_t)BZ * HEADS * (T * DH / 2) * 4);
  u16*   qp     = (u16*)  alloc((size_t)BZ * HEADS * T * NBF * 2);
  u16*   kpT    = (u16*)  alloc((size_t)BZ * HEADS * NBF * T * 2);
  float* xd     = (float*)alloc((size_t)BZ * HEADS * T * NBF * 4);
  float* diag   = (float*)alloc((size_t)BZ * HEADS * T * 4);
  u32*   maxbuf = (u32*)  alloc((size_t)BZ * HEADS * 4);
  float* kpsum  = (float*)alloc((size_t)BZ * HEADS * NBF * 4);
  u16*   ctxb   = (u16*)  alloc((size_t)BZ * HEADS * NBF * DH * 2);
  u32*   ctxpk  = (u32*)  alloc((size_t)BZ * HEADS * (NBF * DH / 2) * 4);
  float* dinv   = (float*)alloc((size_t)BZ * HEADS * T * 4);
  u16*   ob     = (u16*)  alloc((size_t)ROWS * DIM * 2);
  u16*   fb     = (u16*)  alloc((size_t)ROWS * FF * 2);
  u32*   Wqkvp  = (u32*)  alloc((size_t)DEPTH * 3 * (DIM * DIM / 2) * 4);
  u32*   Wop    = (u32*)  alloc((size_t)DEPTH * (DIM * DIM / 2) * 4);
  u32*   W1p    = (u32*)  alloc((size_t)DEPTH * (DIM * FF / 2) * 4);
  u32*   W2p    = (u32*)  alloc((size_t)DEPTH * (FF * DIM / 2) * 4);

  // weight repack: f32 -> packed bf16 WMMA-B fragments (per layer)
  const int dwDD = DIM * DIM / 2;   // 32768 dwords per layer
  const int dwDF = DIM * FF / 2;    // 131072 dwords per layer
  repack_w<<<dim3(dwDD / 256, DEPTH), 256, 0, stream>>>(Wq, Wqkvp, DIM / 16, DIM, dwDD,
                                                        QKV / 16, 0, 3 * dwDD);
  repack_w<<<dim3(dwDD / 256, DEPTH), 256, 0, stream>>>(Wk, Wqkvp, DIM / 16, DIM, dwDD,
                                                        QKV / 16, 16, 3 * dwDD);
  repack_w<<<dim3(dwDD / 256, DEPTH), 256, 0, stream>>>(Wv, Wqkvp, DIM / 16, DIM, dwDD,
                                                        QKV / 16, 32, 3 * dwDD);
  repack_w<<<dim3(dwDD / 256, DEPTH), 256, 0, stream>>>(Wo, Wop, DIM / 16, DIM, dwDD,
                                                        DIM / 16, 0, dwDD);
  repack_w<<<dim3(dwDF / 256, DEPTH), 256, 0, stream>>>(W1, W1p, FF / 16, FF, dwDF,
                                                        FF / 16, 0, dwDF);
  repack_w<<<dim3(dwDF / 256, DEPTH), 256, 0, stream>>>(W2, W2p, DIM / 16, DIM, dwDF,
                                                        DIM / 16, 0, dwDF);

  // x = transpose(mel): (B,DIM,T) -> (B,T,DIM)
  transpose_kernel<<<dim3(T / 32, DIM / 32, BZ), 256, 0, stream>>>(mel, xf);

  const int nBHT = BZ * HEADS * T;
  const int dwV   = T * DH / 2;     // 8192 dwords per (b,h)
  const int dwCtx = NBF * DH / 2;   // 2048 dwords per (b,h)

  // LDS panel bytes per GEMM: (K/32) * NW * 1024
  const int ldsQKV = (DIM / 32) * 4 * 1024;   // 32 KB
  const int ldsCtx = (T / 32)   * 2 * 1024;   // 32 KB
  const int ldsO   = (NBF / 32) * 2 * 1024;   //  8 KB
  const int ldsWo  = (DIM / 32) * 4 * 1024;   // 32 KB
  const int ldsF1  = (DIM / 32) * 4 * 1024;   // 32 KB
  const int ldsF2  = (FF / 32)  * 4 * 1024;   // 128 KB

  for (int l = 0; l < DEPTH; l++) {
    // --- attention ---
    ln_kernel<<<ROWS / 8, 256, 0, stream>>>(xf, ln1g + l * DIM, ln1b + l * DIM, hb);

    // fused qkv = LN(x) @ [Wq|Wk|Wv]   [16384 x 256 x 768]
    gemm_pk<4><<<dim3(QKV / 64, ROWS / 128, 1), 256, ldsQKV, stream>>>(
        hb, Wqkvp + (size_t)l * 3 * dwDD, nullptr, qkv, nullptr, nullptr, nullptr, 0,
        DIM, DIM, QKV / 16, QKV, 1, 0, 0, 0, 0, 0, 0, 0);

    // repack v (per (b,h): T x DH inside fused qkv, ldb=QKV) into B fragments
    repack_b16<<<dim3(dwV / 256, BZ * HEADS), 256, 0, stream>>>(
        qkv + 2 * DIM, vpk, DH / 16, QKV, HEADS,
        (long long)T * QKV, (long long)DH, (long long)dwV, dwV);

    // FAVOR+ features
    qfeat_kernel<<<nBHT / 8, 256, 0, stream>>>(qkv, proj, qp);
    (void)hipMemsetAsync(maxbuf, 0, BZ * HEADS * sizeof(u32), stream);
    kfeat1_kernel<<<nBHT / 8, 256, 0, stream>>>(qkv, proj, xd, diag, maxbuf);
    kfeat2_kernel<<<BZ * HEADS, 128, 0, stream>>>(xd, diag, maxbuf, kpT, kpsum);

    // ctx = kp^T @ v : per (b,h) [128 x 512 x 32]
    gemm_pk<2><<<dim3(1, 1, BZ * HEADS), 256, ldsCtx, stream>>>(
        kpT, vpk, nullptr, ctxb, nullptr, nullptr, nullptr, 0,
        T, T, DH / 16, DH, HEADS,
        (long long)HEADS * NBF * T, (long long)NBF * T,
        (long long)HEADS * dwV, (long long)dwV,
        (long long)HEADS * NBF * DH, (long long)NBF * DH, 0);

    // repack ctx (per (b,h): NBF x DH, ldb=DH)
    repack_b16<<<dim3(dwCtx / 256, BZ * HEADS), 256, 0, stream>>>(
        ctxb, ctxpk, DH / 16, DH, 1,
        (long long)NBF * DH, 0, (long long)dwCtx, dwCtx);

    denom_kernel<<<nBHT / 256, 256, 0, stream>>>(qp, kpsum, dinv);

    // o = (qp @ ctx) * dinv : per (b,h) [512 x 128 x 32] -> (B,T,DIM) slices
    gemm_pk<2><<<dim3(1, T / 128, BZ * HEADS), 256, ldsO, stream>>>(
        qp, ctxpk, nullptr, ob, nullptr, nullptr, dinv, T,
        NBF, NBF, DH / 16, DIM, HEADS,
        (long long)HEADS * T * NBF, (long long)T * NBF,
        (long long)HEADS * dwCtx, (long long)dwCtx,
        (long long)T * DIM, (long long)DH, 0);

    // x = x + o @ Wo + bo
    gemm_pk<4><<<dim3(DIM / 64, ROWS / 128, 1), 256, ldsWo, stream>>>(
        ob, Wop + (size_t)l * dwDD, xf, nullptr, bo + l * DIM, xf, nullptr, 0,
        DIM, DIM, DIM / 16, DIM, 1, 0, 0, 0, 0, 0, 0, 0);

    // --- feedforward ---
    ln_kernel<<<ROWS / 8, 256, 0, stream>>>(xf, ln2g + l * DIM, ln2b + l * DIM, hb);
    gemm_pk<4><<<dim3(FF / 64, ROWS / 128, 1), 256, ldsF1, stream>>>(
        hb, W1p + (size_t)l * dwDF, nullptr, fb, b1 + l * FF, nullptr, nullptr, 0,
        DIM, DIM, FF / 16, FF, 1, 0, 0, 0, 0, 0, 0, 1 /*gelu*/);
    gemm_pk<4><<<dim3(DIM / 64, ROWS / 128, 1), 256, ldsF2, stream>>>(
        fb, W2p + (size_t)l * dwDF, xf, nullptr, b2 + l * DIM, xf, nullptr, 0,
        FF, FF, DIM / 16, DIM, 1, 0, 0, 0, 0, 0, 0, 0);
  }

  mask_kernel<<<(BZ * 4 * T) / 256, 256, 0, stream>>>(xf, Wm, bm, out);
}
